// Layer_74964359185034
// MI455X (gfx1250) — matrix operand order repaired and checked by reading, not compiled, not run on hardware
//
#include <hip/hip_runtime.h>

#define B_ 4
#define S_ 2048
#define D_ 512
#define H_ 8
#define DH_ 64
#define ROWS (B_ * S_) // 8192

typedef __attribute__((ext_vector_type(16))) __bf16 v16bf;
typedef __attribute__((ext_vector_type(8))) __bf16 v8bf;
typedef __attribute__((ext_vector_type(8))) float v8f;
typedef __attribute__((ext_vector_type(4))) unsigned int v4u;
typedef __attribute__((ext_vector_type(8))) int v8i;
typedef __attribute__((ext_vector_type(4))) int v4i;

union AV {
    v16bf v;
    v8bf h[2];
    __bf16 e[16];
    unsigned short u[16];
    unsigned int w[8];
};

#if __has_builtin(__builtin_amdgcn_tensor_load_to_lds) && \
    __has_builtin(__builtin_amdgcn_s_wait_tensorcnt)
#define HAVE_TDM 1
#else
#define HAVE_TDM 0
#endif

__device__ __forceinline__ v8bf ld8(const __bf16* p) {
    return *reinterpret_cast<const v8bf*>(p);
}

// 8 contiguous f32 -> v8bf (v_cvt_pk_bf16_f32, two b128 loads)
__device__ __forceinline__ v8bf cvt8(const float* __restrict__ p) {
    const float4 a = *reinterpret_cast<const float4*>(p);
    const float4 b = *reinterpret_cast<const float4*>(p + 4);
    v8bf r;
    r[0] = (__bf16)a.x; r[1] = (__bf16)a.y; r[2] = (__bf16)a.z; r[3] = (__bf16)a.w;
    r[4] = (__bf16)b.x; r[5] = (__bf16)b.y; r[6] = (__bf16)b.z; r[7] = (__bf16)b.w;
    return r;
}

__device__ __forceinline__ v8f wmma_bf16(const AV& a, const AV& b, v8f c) {
    return __builtin_amdgcn_wmma_f32_16x16x32_bf16(false, a.v, false, b.v,
                                                   (short)0, c, false, false);
}

#if HAVE_TDM
// TDM load of a (rows x 64) bf16 tile (row stride 512 elements, rows packed
// contiguously in LDS). D# per CDNA5 ISA ch.8.
__device__ __forceinline__ void tdm_load_tile64(unsigned lds_off, const __bf16* gptr,
                                                int rows) {
    const unsigned long long ga = (unsigned long long)(uintptr_t)gptr;
    v4u g0 = {1u, lds_off, (unsigned)ga, (unsigned)(ga >> 32) | 0x80000000u};
    v8i g1 = {(int)(1u << 16),          // data_size = 1 (2 bytes), wg_mask = 0
              (int)(64u << 16),         // tensor_dim0 = 64 at bits[63:48]
              (int)((unsigned)rows << 16), // tensor_dim1 at bits[95:80]
              (int)(64u << 16),         // tile_dim0 = 64 at bits[127:112]
              rows,                     // tile_dim1 at bits[143:128]
              512,                      // tensor_dim0_stride lo32 = 512 elems
              0, 0};
    v4i gz = {0, 0, 0, 0};
#if __clang_major__ >= 23
    v8i gz8 = {0, 0, 0, 0, 0, 0, 0, 0};
    __builtin_amdgcn_tensor_load_to_lds(g0, g1, gz, gz, gz8, 0);
#else
    __builtin_amdgcn_tensor_load_to_lds(g0, g1, gz, gz, 0);
#endif
}
#endif

// ---------------- Kernel 1: QKV projection (bf16 WMMA GEMM, 32x32/wave) -----
// q,k stored row-major bf16 [row, n];  v stored TRANSPOSED: vT[(b*512+n)*S + s]
__global__ __launch_bounds__(32) void qkv_proj_kernel(
    const float* __restrict__ embed,
    const float* __restrict__ Wq, const float* __restrict__ bq,
    const float* __restrict__ Wk, const float* __restrict__ bk,
    const float* __restrict__ Wv, const float* __restrict__ bv,
    __bf16* __restrict__ qb, __bf16* __restrict__ kbuf,
    __bf16* __restrict__ vT) {
    const int lane = threadIdx.x;
    const int lo = lane & 15;
    const int hi = lane >> 4;
    const int NT = 48; // 1536 / 32
    const int id = blockIdx.x;
    const int mtile = id / NT;
    const int ntile = id % NT;
    const int which = ntile >> 4; // 0=q 1=k 2=v
    const int ncol0 = (ntile & 15) * 32;

    const float* W = (which == 0) ? Wq : ((which == 1) ? Wk : Wv);
    const float* bias = (which == 0) ? bq : ((which == 1) ? bk : bv);

    const float* a0p = embed + (size_t)(mtile * 32 + lo) * D_;
    const float* a1p = a0p + (size_t)16 * D_;
    const float* w0p = W + (size_t)(ncol0 + lo) * D_;
    const float* w1p = w0p + (size_t)16 * D_;

    v8f c00 = {0.f, 0.f, 0.f, 0.f, 0.f, 0.f, 0.f, 0.f};
    v8f c01 = c00, c10 = c00, c11 = c00;
#pragma unroll 2
    for (int ks = 0; ks < 16; ++ks) {
        const int k0 = ks * 32;
        AV a0, a1, b0, b1;
        a0.h[0] = cvt8(a0p + k0 + hi * 8);      // A elems 0..7 : K = hi*8+0..7
        a0.h[1] = cvt8(a0p + k0 + 16 + hi * 8); // A elems 8..15: K = 16+hi*8+0..7
        a1.h[0] = cvt8(a1p + k0 + hi * 8);
        a1.h[1] = cvt8(a1p + k0 + 16 + hi * 8);
        b0.h[0] = cvt8(w0p + k0 + hi * 8);      // B elems 0..7: K = hi*8+0..7; 8..15: K = 16+hi*8+0..7
        b0.h[1] = cvt8(w0p + k0 + 16 + hi * 8);
        b1.h[0] = cvt8(w1p + k0 + hi * 8);
        b1.h[1] = cvt8(w1p + k0 + 16 + hi * 8);
        c00 = wmma_bf16(a0, b0, c00);
        c01 = wmma_bf16(a0, b1, c01);
        c10 = wmma_bf16(a1, b0, c10);
        c11 = wmma_bf16(a1, b1, c11);
    }
    const float bv0 = bias[ncol0 + lo];
    const float bv1 = bias[ncol0 + 16 + lo];
    const v8f* ct[4] = {&c00, &c01, &c10, &c11};
    if (which < 2) {
        __bf16* outp = (which == 0) ? qb : kbuf;
#pragma unroll
        for (int t = 0; t < 4; ++t) {
            const int ti = t >> 1, tj = t & 1;
            const int col = ncol0 + tj * 16 + lo;
            const float bb = tj ? bv1 : bv0;
#pragma unroll
            for (int r = 0; r < 8; ++r) {
                const int row = mtile * 32 + ti * 16 + r + 8 * hi;
                outp[(size_t)row * D_ + col] = (__bf16)((*ct[t])[r] + bb);
            }
        }
    } else {
#pragma unroll
        for (int t = 0; t < 4; ++t) {
            const int ti = t >> 1, tj = t & 1;
            const int col = ncol0 + tj * 16 + lo;
            const float bb = tj ? bv1 : bv0;
#pragma unroll
            for (int r = 0; r < 8; ++r) {
                const int row = mtile * 32 + ti * 16 + r + 8 * hi;
                const int batch = row >> 11;      // row / S_
                const int ss = row & (S_ - 1);
                vT[((size_t)batch * D_ + col) * S_ + ss] = (__bf16)((*ct[t])[r] + bb);
            }
        }
    }
}

// ---------------- Kernel 2: causal attention (flash-style, WMMA) ------------
// S^T = K*Q^T so P's C-layout equals the A-operand layout for P*V.
// Key tiles processed in PAIRS (32 keys/iter): the two exp'd score tiles fill
// the full 16x32 A operand, and the V B-operand uses both lane halves
// (keys kb0+hi*8 .. +7 and kb0+16+hi*8 .. +7) -> full-rate PV WMMA. Masked (t>s) raw scores are 0
// (NOT -inf) and still enter softmax, matching the reference tril semantics.
// K tiles stream through LDS via the TDM (double-buffered, s_wait_tensorcnt).
__global__ __launch_bounds__(32) void attn_kernel(
    const __bf16* __restrict__ qb, const __bf16* __restrict__ kb,
    const __bf16* __restrict__ vT, float* __restrict__ attn) {
#if HAVE_TDM
    __shared__ unsigned short Kls[2][32 * 64]; // two 4KB K-tile buffers
#endif
    const int lane = threadIdx.x;
    const int lo = lane & 15;
    const int hi = lane >> 4;
    const int QT = S_ / 16;  // 128 query tiles
    const int JP = S_ / 32;  // 64 key-tile pairs
    const int id = blockIdx.x;
    const int itile = id % QT;
    const int bh = id / QT;
    const int h = bh % H_;
    const int b = bh / H_;
    const size_t rowbase = (size_t)b * S_;
    const size_t cbase = (size_t)h * DH_;
    const int q0 = itile * 16;

    // Q as B operand (lane = query col): elems 0..7 = e in ks*32+hi*8+0..7, 8..15 = ks*32+16+hi*8+0..7
    AV qB[2];
    {
        const __bf16* qp = qb + (rowbase + q0 + lo) * D_ + cbase;
#pragma unroll
        for (int ks = 0; ks < 2; ++ks) {
            qB[ks].h[0] = ld8(qp + ks * 32 + hi * 8);
            qB[ks].h[1] = ld8(qp + ks * 32 + 16 + hi * 8);
        }
    }

    v8f o[4];
#pragma unroll
    for (int nt = 0; nt < 4; ++nt)
        o[nt] = (v8f){0.f, 0.f, 0.f, 0.f, 0.f, 0.f, 0.f, 0.f};
    float m = -1e30f, lsum = 0.f;
    const float scale = 0.044194173824159216f; // 1/sqrt(512)

    const __bf16* kbase = kb + rowbase * D_ + cbase; // key row t at + t*D_
    const __bf16* vbase = vT + ((size_t)b * D_ + cbase) * S_;

#if HAVE_TDM
    tdm_load_tile64((unsigned)(uintptr_t)&Kls[0][0], kbase, 32);
#endif

    for (int j = 0; j < JP; ++j) {
        const int kb0 = j * 32; // first key of this pair
#if HAVE_TDM
        if (j + 1 < JP) {
            tdm_load_tile64((unsigned)(uintptr_t)&Kls[(j + 1) & 1][0],
                            kbase + (size_t)(j + 1) * 32 * D_, 32);
            __builtin_amdgcn_s_wait_tensorcnt((short)1); // pair j complete
        } else {
            __builtin_amdgcn_s_wait_tensorcnt((short)0);
        }
        asm volatile("" ::: "memory"); // keep LDS reads after the wait
        const __bf16* kp0 = (const __bf16*)&Kls[j & 1][lo * 64];
        const __bf16* kp1 = (const __bf16*)&Kls[j & 1][(16 + lo) * 64];
#else
        const __bf16* kp0 = kbase + (size_t)(kb0 + lo) * D_;
        const __bf16* kp1 = kp0 + (size_t)16 * D_;
#endif
        // prefetch upcoming V columns (global_prefetch_b8)
        __builtin_prefetch((const void*)(vbase + (size_t)lo * S_ + kb0 + 64), 0, 0);

        // --- two S^T tiles: A = K rows (lane = key), B = Q ---
        v8f c0 = {0.f, 0.f, 0.f, 0.f, 0.f, 0.f, 0.f, 0.f};
        v8f c1 = c0;
#pragma unroll
        for (int ks = 0; ks < 2; ++ks) {
            AV ka0, ka1;
            ka0.h[0] = ld8(kp0 + ks * 32 + hi * 8);
            ka0.h[1] = ld8(kp0 + ks * 32 + 16 + hi * 8);
            ka1.h[0] = ld8(kp1 + ks * 32 + hi * 8);
            ka1.h[1] = ld8(kp1 + ks * 32 + 16 + hi * 8);
            c0 = wmma_bf16(ka0, qB[ks], c0);
            c1 = wmma_bf16(ka1, qB[ks], c1);
        }
        // mask (tril on raw scores -> 0) + scale; C: M=key, N=query=lo
        const int squery = q0 + lo;
#pragma unroll
        for (int r = 0; r < 8; ++r) {
            const int t0 = kb0 + r + 8 * hi;
            const int t1 = t0 + 16;
            c0[r] = (t0 > squery) ? 0.f : c0[r] * scale;
            c1[r] = (t1 > squery) ? 0.f : c1[r] * scale;
        }
        // --- online softmax stats per query (per lane column) ---
        float tm = fmaxf(c0[0], c1[0]);
#pragma unroll
        for (int r = 1; r < 8; ++r) tm = fmaxf(tm, fmaxf(c0[r], c1[r]));
        tm = fmaxf(tm, __shfl_xor(tm, 16, 32));
        const float mnew = fmaxf(m, tm);
        const float alpha = __expf(m - mnew);
        AV pA;
        float ps = 0.f;
#pragma unroll
        for (int r = 0; r < 8; ++r) {
            const float e0 = __expf(c0[r] - mnew);
            const float e1 = __expf(c1[r] - mnew);
            ps += e0 + e1;
            pA.e[r] = (__bf16)e0;     // A elems 0..7  = keys kb0 + (hi?8..15:0..7)
            pA.e[r + 8] = (__bf16)e1; // A elems 8..15 = keys kb0+16 + (...)
        }
        ps += __shfl_xor(ps, 16, 32);
        lsum = lsum * alpha + ps;
        m = mnew;
        // rescale accumulators: O row M = r + 8*hi -> alpha of that query
        float a8[8];
#pragma unroll
        for (int r = 0; r < 8; ++r) a8[r] = __shfl(alpha, r + 8 * hi, 32);
#pragma unroll
        for (int nt = 0; nt < 4; ++nt)
#pragma unroll
            for (int r = 0; r < 8; ++r) o[nt][r] *= a8[r];

        // --- O += P * V ; B operand from pre-transposed vT, full 32-deep K ---
        // lane = e col; elems 0..7 = keys kb0 + hi*8 + 0..7, 8..15 = kb0 + 16 + hi*8 + 0..7.
#pragma unroll
        for (int nt = 0; nt < 4; ++nt) {
            const __bf16* vp = vbase + (size_t)(nt * 16 + lo) * S_ + kb0 + hi * 8;
            AV vB;
            vB.h[0] = ld8(vp);
            vB.h[1] = ld8(vp + 16);
            o[nt] = wmma_bf16(pA, vB, o[nt]);
        }
    }

    // normalize by row sums and store f32 [B,S,D] (col = h*64 + e)
    float linv8[8];
#pragma unroll
    for (int r = 0; r < 8; ++r) linv8[r] = 1.0f / __shfl(lsum, r + 8 * hi, 32);
#pragma unroll
    for (int nt = 0; nt < 4; ++nt)
#pragma unroll
        for (int r = 0; r < 8; ++r) {
            const int srow = q0 + r + 8 * hi;
            attn[(rowbase + srow) * D_ + cbase + nt * 16 + lo] = o[nt][r] * linv8[r];
        }
}

// ---------------- Kernel 3: output projection + residual (32x32/wave) -------
__global__ __launch_bounds__(32) void oproj_kernel(
    const float* __restrict__ attn, const float* __restrict__ Wo,
    const float* __restrict__ bo, const float* __restrict__ embed,
    float* __restrict__ xout) {
    const int lane = threadIdx.x;
    const int lo = lane & 15;
    const int hi = lane >> 4;
    const int NT = 16; // 512 / 32
    const int id = blockIdx.x;
    const int mtile = id / NT;
    const int ntile = id % NT;
    const int ncol0 = ntile * 32;

    const float* a0p = attn + (size_t)(mtile * 32 + lo) * D_;
    const float* a1p = a0p + (size_t)16 * D_;
    const float* w0p = Wo + (size_t)(ncol0 + lo) * D_;
    const float* w1p = w0p + (size_t)16 * D_;

    v8f c00 = {0.f, 0.f, 0.f, 0.f, 0.f, 0.f, 0.f, 0.f};
    v8f c01 = c00, c10 = c00, c11 = c00;
#pragma unroll 2
    for (int ks = 0; ks < 16; ++ks) {
        const int k0 = ks * 32;
        AV a0, a1, b0, b1;
        a0.h[0] = cvt8(a0p + k0 + hi * 8);
        a0.h[1] = cvt8(a0p + k0 + 16 + hi * 8);
        a1.h[0] = cvt8(a1p + k0 + hi * 8);
        a1.h[1] = cvt8(a1p + k0 + 16 + hi * 8);
        b0.h[0] = cvt8(w0p + k0 + hi * 8);
        b0.h[1] = cvt8(w0p + k0 + 16 + hi * 8);
        b1.h[0] = cvt8(w1p + k0 + hi * 8);
        b1.h[1] = cvt8(w1p + k0 + 16 + hi * 8);
        c00 = wmma_bf16(a0, b0, c00);
        c01 = wmma_bf16(a0, b1, c01);
        c10 = wmma_bf16(a1, b0, c10);
        c11 = wmma_bf16(a1, b1, c11);
    }
    const float bv0 = bo[ncol0 + lo];
    const float bv1 = bo[ncol0 + 16 + lo];
    const v8f* ct[4] = {&c00, &c01, &c10, &c11};
#pragma unroll
    for (int t = 0; t < 4; ++t) {
        const int ti = t >> 1, tj = t & 1;
        const int col = ncol0 + tj * 16 + lo;
        const float bb = tj ? bv1 : bv0;
#pragma unroll
        for (int r = 0; r < 8; ++r) {
            const int row = mtile * 32 + ti * 16 + r + 8 * hi;
            xout[(size_t)row * D_ + col] =
                (*ct[t])[r] + bb + embed[(size_t)row * D_ + col];
        }
    }
}

// ---------------- Kernel 4: LayerNorm over D=512 ----------------------------
__global__ __launch_bounds__(256) void ln_kernel(
    const float* __restrict__ x, const float* __restrict__ gamma,
    const float* __restrict__ beta, float* __restrict__ out) {
    __shared__ float s1[256], s2[256];
    const int row = blockIdx.x;
    const int t = threadIdx.x;
    const float* xr = x + (size_t)row * D_;
    const float v0 = xr[t], v1 = xr[t + 256];
    s1[t] = v0 + v1;
    s2[t] = v0 * v0 + v1 * v1;
    __syncthreads();
    for (int off = 128; off > 0; off >>= 1) {
        if (t < off) {
            s1[t] += s1[t + off];
            s2[t] += s2[t + off];
        }
        __syncthreads();
    }
    const float mu = s1[0] * (1.0f / 512.0f);
    const float var = s2[0] * (1.0f / 512.0f) - mu * mu;
    const float rstd = rsqrtf(var + 1e-5f);
    out[(size_t)row * D_ + t] = (v0 - mu) * rstd * gamma[t] + beta[t];
    out[(size_t)row * D_ + t + 256] = (v1 - mu) * rstd * gamma[t + 256] + beta[t + 256];
}

extern "C" void kernel_launch(void* const* d_in, const int* in_sizes, int n_in,
                              void* d_out, int out_size, void* d_ws, size_t ws_size,
                              hipStream_t stream) {
    (void)in_sizes; (void)n_in; (void)out_size; (void)ws_size;
    const float* embed = (const float*)d_in[0];
    const float* Wq = (const float*)d_in[1];
    const float* bq = (const float*)d_in[2];
    const float* Wk = (const float*)d_in[3];
    const float* bk = (const float*)d_in[4];
    const float* Wv = (const float*)d_in[5];
    const float* bv = (const float*)d_in[6];
    const float* Wo = (const float*)d_in[7];
    const float* bo = (const float*)d_in[8];
    const float* gamma = (const float*)d_in[9];
    const float* beta = (const float*)d_in[10];

    char* w = (char*)d_ws;
    const size_t QS = (size_t)ROWS * D_ * sizeof(unsigned short); // 8 MiB each
    __bf16* qbuf = (__bf16*)(w);
    __bf16* kbuf = (__bf16*)(w + QS);
    __bf16* vTbuf = (__bf16*)(w + 2 * QS);
    float* attn = (float*)(w + 3 * QS);
    float* xbuf = (float*)(w + 3 * QS + (size_t)ROWS * D_ * sizeof(float));

    qkv_proj_kernel<<<(ROWS / 32) * 48, 32, 0, stream>>>(embed, Wq, bq, Wk, bk, Wv, bv,
                                                         qbuf, kbuf, vTbuf);
    attn_kernel<<<B_ * H_ * (S_ / 16), 32, 0, stream>>>(qbuf, kbuf, vTbuf, attn);
    oproj_kernel<<<(ROWS / 32) * 16, 32, 0, stream>>>(attn, Wo, bo, embed, xbuf);
    ln_kernel<<<ROWS, 256, 0, stream>>>(xbuf, gamma, beta, (float*)d_out);
}
